// BaselineCapsNet_43903155699883
// MI455X (gfx1250) — matrix operand order repaired and checked by reading, not compiled, not run on hardware
//
#include <hip/hip_runtime.h>

// ---------------------------------------------------------------------------
// CapsNet forward for MI455X (gfx1250, wave32).
// Hotspot (primary-caps conv, ~98 GFLOP) runs as implicit GEMM on
// v_wmma_f32_16x16x32_bf16 with NHWC activations; tiles are staged into
// double-buffered LDS with global_load_async_to_lds_b128 (ASYNCcnt-tracked
// CDNA5 async DMA). The K loop is kept un-unrolled with pointer-toggled
// buffers so accumulators stay pinned (no v_mov/v_nop hazard trains).
// u_hat runs on v_wmma_f32_16x16x4_f32 (fp32, K=8). Routing is
// bandwidth-bound VALU work (3x 189MB u_hat sweeps ~ 25us @ 23.3 TB/s).
// ---------------------------------------------------------------------------

typedef __attribute__((ext_vector_type(16))) __bf16        v16bf;
typedef __attribute__((ext_vector_type(8)))  float         v8f;
typedef __attribute__((ext_vector_type(2)))  float         v2f;
typedef __attribute__((ext_vector_type(4)))  unsigned int  v4u;

union AFrag { v16bf v; v4u q[2]; };

__device__ __forceinline__ unsigned short f2bf(float f) {
  unsigned int x = __float_as_uint(f);
  unsigned int r = x + 0x7FFFu + ((x >> 16) & 1u);   // round-to-nearest-even
  return (unsigned short)(r >> 16);
}

// 32 contiguous bytes: global -> LDS via async DMA (no VGPR data round-trip).
// Same INST_OFFSET is applied to both LDS and memory address (ISA 10.x).
__device__ __forceinline__ void async_copy32(unsigned lds, const void* g) {
  asm volatile(
      "global_load_async_to_lds_b128 %0, %1, off\n\t"
      "global_load_async_to_lds_b128 %0, %1, off offset:16"
      :: "v"(lds), "v"(g) : "memory");
}
__device__ __forceinline__ void wait_async0() {
  asm volatile("s_wait_asynccnt 0x0" ::: "memory");
}

// ---------------------------------------------------------------------------
// Weights f32 -> bf16 with K permuted to k' = (ky*9+kx)*256 + ic so that a
// 32-wide K chunk of the B (activation) matrix is contiguous in NHWC memory.
// wbf[m][kk*256+ic] = prim_w[m][ic*81+kk].  Contiguous writes, L2 gathers.
// ---------------------------------------------------------------------------
__global__ void k_wcvt(const float* __restrict__ w, unsigned short* __restrict__ wbf, int n) {
  int i = blockIdx.x * 256 + threadIdx.x;
  if (i >= n) return;
  const int m  = i / 20736;
  const int kp = i - m * 20736;
  const int kk = kp >> 8;      // ky*9+kx  (0..80)
  const int ic = kp & 255;
  wbf[i] = f2bf(w[m * 20736 + ic * 81 + kk]);
}

// ---------------------------------------------------------------------------
// conv1: [256,1,28,28] -> relu -> bf16 NHWC [256][20][20][256].
// threads = oc (256), block = (4 output positions, b): filter taps in
// registers, image in LDS, stores fully coalesced (256 consecutive oc).
// ---------------------------------------------------------------------------
__global__ __launch_bounds__(256) void k_conv1(
    const float* __restrict__ img, const float* __restrict__ w,
    const float* __restrict__ bias, unsigned short* __restrict__ xbf) {
  const int b    = blockIdx.y;
  const int pos0 = blockIdx.x * 4;
  const int oc   = threadIdx.x;
  __shared__ float im[784];
  for (int i = oc; i < 784; i += 256) im[i] = img[b * 784 + i];
  __syncthreads();
  float wr[81];
#pragma unroll
  for (int t = 0; t < 81; ++t) wr[t] = w[oc * 81 + t];
  const float bv = bias[oc];
#pragma unroll
  for (int q = 0; q < 4; ++q) {
    const int pos = pos0 + q;
    const int oy = pos / 20, ox = pos - oy * 20;
    float s = bv;
#pragma unroll
    for (int ky = 0; ky < 9; ++ky)
#pragma unroll
      for (int kx = 0; kx < 9; ++kx)
        s = fmaf(wr[ky * 9 + kx], im[(oy + ky) * 28 + (ox + kx)], s);
    s = fmaxf(s, 0.f);
    xbf[(b * 400 + pos) * 256 + oc] = f2bf(s);
  }
}

// ---------------------------------------------------------------------------
// Primary caps implicit GEMM, WMMA bf16.
//   C[m,n] = sum_k' A[m,k'] * B[k',n]
//   m = out-channel (256), n = b*36+p (9216), k' = (ky*9+kx)*256 + ic (20736)
// NHWC activations => each thread stages 32 contiguous bytes of A and of B
// per stage via async-to-LDS DMA into double-buffered LDS. Buffer parity is
// carried by sign-toggled pointer deltas so the K loop has a single body and
// accumulators never move. 8 waves: 2(M) x 4(N), each wave 4x2 16x16 tiles.
// Output scattered into u[B][8][1152] (+bias).
// ---------------------------------------------------------------------------
__global__ __launch_bounds__(256) void k_conv2_gemm(
    const unsigned short* __restrict__ wbf,   // [256][20736] bf16, k' order
    const unsigned short* __restrict__ xbf,   // [256][20][20][256] bf16 NHWC
    const float* __restrict__ prim_b,         // [256]
    float* __restrict__ u)                    // [256][8][1152]
{
  __shared__ __align__(16) unsigned short As[2 * 128 * 40];  // [m][k], 16B rows
  __shared__ __align__(16) unsigned short Bs[2 * 128 * 40];  // [n][k]
  constexpr int kBufB = 128 * 40 * 2;   // bytes per buffer
  constexpr int kBufE = 128 * 40;       // elements per buffer

  const int tid  = threadIdx.x;
  const int lane = tid & 31;
  const int wid  = tid >> 5;
  const int wm   = wid & 1;        // 2 M-waves (64 rows)
  const int wn   = wid >> 1;       // 4 N-waves (32 cols)
  const int nB   = blockIdx.x * 128;
  const int mB   = blockIdx.y * 128;
  const int hi   = lane >> 4;      // lane half selects K sub-ranges (ISA 7.12.2)
  const int l16  = lane & 15;

  // A staging: thread -> (row, 16 consecutive k'), 32B contiguous
  const int a_row = tid >> 1;
  const int a_col = (tid & 1) * 16;
  const unsigned short* aGp = wbf + (size_t)(mB + a_row) * 20736 + a_col;

  // B staging: thread -> (one n column, 16 consecutive ic), 32B contiguous
  const int b_nl = tid >> 1;       // 0..127
  const int b_h  = (tid & 1) * 16; // ic half
  int xbase;
  {
    const int n  = nB + b_nl;
    const int bb = n / 36;
    const int p  = n - bb * 36;
    const int oy = p / 6, ox = p - oy * 6;
    // NHWC: addr = b*102400 + iy*5120 + ix*256 + ic, iy=2oy+ky, ix=2ox+kx
    xbase = bb * 102400 + oy * 10240 + ox * 512 + b_h;
  }

  // staging LDS destinations (32-bit DS addresses) and fragment pointers
  unsigned aSt = (unsigned)(size_t)&As[a_row * 40 + a_col];
  unsigned bSt = (unsigned)(size_t)&Bs[b_nl * 40 + b_h];
  const unsigned short* aFp = &As[(wm * 64 + l16) * 40 + (hi ? 8 : 0)];
  const unsigned short* bFp = &Bs[(wn * 32 + l16) * 40 + (hi ? 8 : 0)];

  const v8f vz = {0.f, 0.f, 0.f, 0.f, 0.f, 0.f, 0.f, 0.f};
  v8f acc[4][2];
#pragma unroll
  for (int mt = 0; mt < 4; ++mt)
#pragma unroll
    for (int nt = 0; nt < 2; ++nt) acc[mt][nt] = vz;

  // one K-step (32) of WMMAs from the current fragment pointers
  auto compute = [&]() {
    AFrag bfr0, bfr1;
    // lanes 0-15: K{0..7},{16..23}; lanes 16-31: K{8..15},{24..31}
    bfr0.q[0] = *(const v4u*)(bFp);
    bfr0.q[1] = *(const v4u*)(bFp + 16);
    bfr1.q[0] = *(const v4u*)(bFp + 640);
    bfr1.q[1] = *(const v4u*)(bFp + 656);
#pragma unroll
    for (int mt = 0; mt < 4; ++mt) {
      AFrag af;
      // lanes 0-15: K{0..7},{16..23}; lanes 16-31: K{8..15},{24..31}
      af.q[0] = *(const v4u*)(aFp + mt * 640);
      af.q[1] = *(const v4u*)(aFp + mt * 640 + 16);
      acc[mt][0] = __builtin_amdgcn_wmma_f32_16x16x32_bf16(
          false, af.v, false, bfr0.v, (short)0, acc[mt][0], false, false);
      acc[mt][1] = __builtin_amdgcn_wmma_f32_16x16x32_bf16(
          false, af.v, false, bfr1.v, (short)0, acc[mt][1], false, false);
    }
  };

  // prologue: stage K-chunk 0 into buffer 0
  async_copy32(aSt, aGp);
  async_copy32(bSt, xbf + xbase);          // k0=0 -> ky=kx=ic0=0
  aSt += kBufB; bSt += kBufB;              // next staging -> buffer 1
  aGp += 32;
  wait_async0();
  __syncthreads();

  int fd = kBufE;        // fragment-pointer toggle (elements)
  int sd = -kBufB;       // staging-address toggle (bytes)
  int k0n = 32;          // K-chunk being staged this iteration (uniform)

#pragma clang loop unroll(disable)
  for (int s = 0; s < 647; ++s) {
    {  // stage K-chunk k0n into the other buffer (async DMA)
      const int kk  = k0n >> 8;            // ky*9+kx (wave-uniform -> SALU)
      const int ic0 = k0n & 255;
      const int ky  = kk / 9, kx = kk - ky * 9;
      async_copy32(aSt, aGp);
      async_copy32(bSt, xbf + xbase + ky * 5120 + kx * 256 + ic0);
    }
    compute();
    wait_async0();       // staged tile fully in LDS before publishing
    __syncthreads();
    aFp += fd; bFp += fd; fd = -fd;
    aSt += sd; bSt += sd; sd = -sd;
    aGp += 32; k0n += 32;
  }
  compute();             // final K-chunk (no staging)

  // writeback: C(m,n) -> u[b][i][r], r = mm*36+p, m = i*32+mm
  const int mo = (lane & 16) ? 8 : 0;   // C layout: lanes 16-31 hold M+8
#pragma unroll
  for (int nt = 0; nt < 2; ++nt) {
    const int n  = nB + wn * 32 + nt * 16 + l16;
    const int bb = n / 36;
    const int p  = n - bb * 36;
#pragma unroll
    for (int mt = 0; mt < 4; ++mt) {
#pragma unroll
      for (int v = 0; v < 8; ++v) {
        const int m  = mB + wm * 64 + mt * 16 + v + mo;
        const int i  = m >> 5, mm = m & 31;
        u[(bb * 8 + i) * 1152 + mm * 36 + p] = acc[mt][nt][v] + prim_b[m];
      }
    }
  }
}

// ---------------------------------------------------------------------------
// squash over the ROUTES axis (reference squashes dim=1 of [B,1152,8]).
// u layout [B][8][1152] -> contiguous reductions, in place. Block per (b,i).
// ---------------------------------------------------------------------------
__global__ __launch_bounds__(256) void k_usquash(float* __restrict__ u) {
  const int tid = threadIdx.x;
  float* base = u + (size_t)blockIdx.x * 1152;
  __shared__ float red[256];
  float s = 0.f;
  for (int r = tid; r < 1152; r += 256) { float x = base[r]; s += x * x; }
  red[tid] = s;
  __syncthreads();
  for (int off = 128; off > 0; off >>= 1) {
    if (tid < off) red[tid] += red[tid + off];
    __syncthreads();
  }
  const float sq = red[0];
  const float scale = sq / (1.f + sq) * rsqrtf(sq);
  for (int r = tid; r < 1152; r += 256) base[r] *= scale;
}

// ---------------------------------------------------------------------------
// u_hat[b,r,j,d] = sum_i W[r,j,d,i] * u[b,i,r] on v_wmma_f32_16x16x4_f32.
// One wave per (route r, 16-batch tile): M=16 b, N=160 jd, K=8 (2 K-steps).
// A (u) per ISA 32-bit 16x4 layout: lanes 0-15 K{0,1}, lanes 16-31 K{2,3}.
// ---------------------------------------------------------------------------
__global__ __launch_bounds__(32) void k_uhat(
    const float* __restrict__ W, const float* __restrict__ u,
    float* __restrict__ uh) {
  const int r   = blockIdx.x;
  const int b0  = blockIdx.y * 16;
  const int lane = threadIdx.x;
  const int hi  = lane >> 4;
  const int l16 = lane & 15;

  const v8f vz = {0.f, 0.f, 0.f, 0.f, 0.f, 0.f, 0.f, 0.f};
  v8f acc[10];
#pragma unroll
  for (int nt = 0; nt < 10; ++nt) acc[nt] = vz;

  const float* Wr = W + (size_t)r * 1280;
#pragma unroll
  for (int ks = 0; ks < 2; ++ks) {
    const int i0 = ks * 4 + hi * 2;
    v2f a;
    a.x = u[(size_t)((b0 + l16) * 8 + i0) * 1152 + r];
    a.y = u[(size_t)((b0 + l16) * 8 + i0 + 1) * 1152 + r];
#pragma unroll
    for (int nt = 0; nt < 10; ++nt) {
      const v2f bb = *(const v2f*)(Wr + (nt * 16 + l16) * 8 + i0);
      acc[nt] = __builtin_amdgcn_wmma_f32_16x16x4_f32(
          false, a, false, bb, (short)0, acc[nt], false, false);
    }
  }

  const int mo = hi ? 8 : 0;
#pragma unroll
  for (int nt = 0; nt < 10; ++nt) {
    const int jd = nt * 16 + l16;
#pragma unroll
    for (int v = 0; v < 8; ++v) {
      const int b = b0 + v + mo;
      uh[((size_t)b * 1152 + r) * 160 + jd] = acc[nt][v];
    }
  }
}

__global__ void k_zero(float* __restrict__ p, int n) {
  int i = blockIdx.x * 256 + threadIdx.x;
  if (i < n) p[i] = 0.f;
}

// c[r,:] = softmax over the 10 capsules of b_ij[r,:]
__global__ void k_softmax(const float* __restrict__ bij, float* __restrict__ cij) {
  int r = blockIdx.x * 256 + threadIdx.x;
  if (r >= 1152) return;
  float x[10], m = -1e30f;
#pragma unroll
  for (int j = 0; j < 10; ++j) { x[j] = bij[r * 10 + j]; m = fmaxf(m, x[j]); }
  float s = 0.f;
#pragma unroll
  for (int j = 0; j < 10; ++j) { x[j] = __expf(x[j] - m); s += x[j]; }
  const float inv = 1.f / s;
#pragma unroll
  for (int j = 0; j < 10; ++j) cij[r * 10 + j] = x[j] * inv;
}

// s_j[b,j,d] = sum_r c[r,j]*u_hat[b,r,j,d]; v = squash(s over d).
// Block per b; 160 threads = (j,d); c staged in LDS (46 KB << 320 KB pool).
__global__ __launch_bounds__(160) void k_sv(
    const float* __restrict__ cij, const float* __restrict__ uh,
    float* __restrict__ v, float* __restrict__ out) {
  const int b = blockIdx.x;
  const int tid = threadIdx.x;
  const int j = tid >> 4;
  __shared__ float cl[11520];
  __shared__ float red[160];
  for (int i = tid; i < 11520; i += 160) cl[i] = cij[i];
  __syncthreads();
  const float* uhb = uh + (size_t)b * 1152 * 160;
  float s = 0.f;
#pragma unroll 4
  for (int r = 0; r < 1152; ++r)
    s = fmaf(cl[r * 10 + j], uhb[r * 160 + tid], s);
  red[tid] = s * s;
  __syncthreads();
  float sq = 0.f;
#pragma unroll
  for (int d = 0; d < 16; ++d) sq += red[j * 16 + d];
  const float scale = sq / (1.f + sq) * rsqrtf(sq);
  const float vv = s * scale;
  v[b * 160 + tid] = vv;
  out[b * 160 + tid] = vv;
}

// b_ij[r,j] += mean_b sum_d u_hat[b,r,j,d]*v[b,j,d]. Block per route.
__global__ __launch_bounds__(160) void k_agree(
    const float* __restrict__ uh, const float* __restrict__ v,
    float* __restrict__ bij) {
  const int r = blockIdx.x;
  const int tid = threadIdx.x;
  __shared__ float red[160];
  float acc = 0.f;
#pragma unroll 4
  for (int b = 0; b < 256; ++b)
    acc = fmaf(uh[((size_t)b * 1152 + r) * 160 + tid], v[b * 160 + tid], acc);
  red[tid] = acc;
  __syncthreads();
  if (tid < 10) {
    float s = 0.f;
#pragma unroll
    for (int d = 0; d < 16; ++d) s += red[tid * 16 + d];
    bij[r * 10 + tid] += s * (1.f / 256.f);
  }
}

// ---------------------------------------------------------------------------
extern "C" void kernel_launch(void* const* d_in, const int* in_sizes, int n_in,
                              void* d_out, int out_size, void* d_ws, size_t ws_size,
                              hipStream_t stream) {
  const float* images  = (const float*)d_in[0];
  // d_in[1] labels: unused by forward
  const float* conv1_w = (const float*)d_in[2];
  const float* conv1_b = (const float*)d_in[3];
  const float* prim_w  = (const float*)d_in[4];
  const float* prim_b  = (const float*)d_in[5];
  const float* W       = (const float*)d_in[6];
  float* out = (float*)d_out;

  char* ws = (char*)d_ws;
  size_t off = 0;
  auto take = [&](size_t bytes) -> char* {
    char* p = ws + off;
    off = (off + bytes + 255) & ~(size_t)255;
    return p;
  };
  unsigned short* xbf = (unsigned short*)take(256ull * 400 * 256 * 2);  // 52.4 MB NHWC
  unsigned short* wbf = (unsigned short*)take(256ull * 20736 * 2);      // 10.6 MB
  float* u   = (float*)take(256ull * 8 * 1152 * 4);                     //  9.4 MB
  float* uh  = (float*)take(256ull * 1152 * 160 * 4);                   // 188.7 MB
  float* bij = (float*)take(1152 * 10 * 4);
  float* cij = (float*)take(1152 * 10 * 4);
  float* vv  = (float*)take(256 * 160 * 4);

  k_wcvt<<<20736, 256, 0, stream>>>(prim_w, wbf, 256 * 20736);
  k_conv1<<<dim3(100, 256), 256, 0, stream>>>(images, conv1_w, conv1_b, xbf);
  k_conv2_gemm<<<dim3(72, 2), 256, 0, stream>>>(wbf, xbf, prim_b, u);
  k_usquash<<<2048, 256, 0, stream>>>(u);
  k_uhat<<<dim3(1152, 16), 32, 0, stream>>>(W, u, uh);
  k_zero<<<45, 256, 0, stream>>>(bij, 11520);
  for (int t = 0; t < 3; ++t) {
    k_softmax<<<5, 256, 0, stream>>>(bij, cij);
    k_sv<<<256, 160, 0, stream>>>(cij, uh, vv, out);
    if (t < 2) k_agree<<<1152, 160, 0, stream>>>(uh, vv, bij);
  }
}
